// EncoderLayer_34565896798547
// MI455X (gfx1250) — matrix operand order repaired and checked by reading, not compiled, not run on hardware
//
#include <hip/hip_runtime.h>
#include <hip/hip_bf16.h>

// ---------------------------------------------------------------------------
// CDNA5 (gfx1250) transformer encoder layer.
// All GEMMs via v_wmma_f32_16x16x32_f16 (wave32, f16 in / f32 accumulate),
// double-buffered LDS, 4 or 8 WMMA tiles per wave per K-step.
// ---------------------------------------------------------------------------

typedef __attribute__((ext_vector_type(16))) _Float16 v16h;
typedef __attribute__((ext_vector_type(8)))  _Float16 v8h;
typedef __attribute__((ext_vector_type(8)))  float    v8f;

#define DM     1024          // d_model
#define NH     16            // heads
#define DKH    64            // d_k per head
#define DFF    4096          // ffn hidden
#define SEQ    2048          // sequence length
#define BBATCH 2             // batch
#define MROWS  (BBATCH*SEQ)  // 4096 token rows
#define NBZ    (BBATCH*NH)   // 32 attention batches

// ------------------------- elementwise f32 -> f16 --------------------------
__global__ void cvt_f32_f16_kernel(const float* __restrict__ in,
                                   _Float16* __restrict__ out, int total) {
  int i = blockIdx.x * 256 + threadIdx.x;
  if (i < total) out[i] = (_Float16)in[i];
}

// ------------------- W[K][N] (fp32) -> Wt[N][K] (f16) -----------------------
__global__ void transpose_f32_f16_kernel(const float* __restrict__ W,
                                         _Float16* __restrict__ Wt,
                                         int K, int N, int total) {
  int i = blockIdx.x * 256 + threadIdx.x;
  if (i >= total) return;
  int n = i / K;
  int k = i - n * K;
  Wt[i] = (_Float16)W[(size_t)k * N + n];
}

// ---------------------------------------------------------------------------
// Generic batched GEMM:  C = act( scale * A[M,K] * Bt[N,K]^T + bias[n] )
// 256 threads = 8 wave32s, arranged WR x WC; each wave owns TM x TN WMMA
// tiles (16x16 each).  Block tile: BM = WR*TM*16, BN = WC*TN*16, BK = 32.
// LDS double-buffered: one barrier per K-step; next tile's global loads are
// issued before the WMMAs so their latency hides under matrix work.
// Output index fully generic: idx = zoff +
//     (m/msplit)*pm0 + (m%msplit)*pm1 + (n/nsplit)*pn0 + (n%nsplit)*pn1
// ---------------------------------------------------------------------------
#define BK  32
#define SLD 40   // LDS row stride in halfs (80 bytes, 16B aligned, padded)

template<typename OutT, int TM, int TN, int WR, int WC>
__global__ __launch_bounds__(256)
void gemm_bt_kernel(const _Float16* __restrict__ A, size_t zStrideA, int lda,
                    const _Float16* __restrict__ Bt, size_t zStrideB, int ldb,
                    OutT* __restrict__ C, int zsplit, size_t pz0, size_t pz1,
                    int K, const float* __restrict__ bias, float scale, int relu,
                    int msplit, size_t pm0, size_t pm1,
                    int nsplit, size_t pn0, size_t pn1)
{
  constexpr int BM  = WR * TM * 16;
  constexpr int BN  = WC * TN * 16;
  constexpr int ACH = (BM * BK / 8) / 256;   // v8h chunks per thread (A tile)
  constexpr int BCH = (BN * BK / 8) / 256;   // v8h chunks per thread (B tile)

  __shared__ __align__(16) _Float16 sA[2][BM * SLD];
  __shared__ __align__(16) _Float16 sB[2][BN * SLD];

  const int tid  = threadIdx.x;
  const int lane = tid & 31;
  const int wave = tid >> 5;        // 0..7
  const int wm   = wave % WR;       // wave row
  const int wn   = wave / WR;       // wave col
  const int lr   = lane & 15;       // row/col within 16
  const int lg   = lane >> 4;       // lane half-group

  const int z  = blockIdx.z;
  const int m0 = blockIdx.y * BM;
  const int n0 = blockIdx.x * BN;

  const _Float16* Az = A  + (size_t)z * zStrideA;
  const _Float16* Bz = Bt + (size_t)z * zStrideB;

  v8f acc[TM][TN] = {};

  // ---- prologue: stage K-step 0 into buffer 0 ----
  {
#pragma unroll
    for (int j = 0; j < ACH; ++j) {
      const int ci = tid + j * 256, row = ci >> 2, c8 = (ci & 3) * 8;
      *(v8h*)&sA[0][row * SLD + c8] =
          *(const v8h*)&Az[(size_t)(m0 + row) * lda + c8];
    }
#pragma unroll
    for (int j = 0; j < BCH; ++j) {
      const int ci = tid + j * 256, row = ci >> 2, c8 = (ci & 3) * 8;
      *(v8h*)&sB[0][row * SLD + c8] =
          *(const v8h*)&Bz[(size_t)(n0 + row) * ldb + c8];
    }
  }

  int cur = 0;
  for (int k0 = 0; k0 < K; k0 += BK) {
    __syncthreads();   // staged tile visible; prior reads of other buf done
    const int kn = k0 + BK;
    const bool hasNext = kn < K;

    // ---- issue next tile's global loads early (latency under WMMAs) ----
    v8h ra[ACH], rb[BCH];
    if (hasNext) {
#pragma unroll
      for (int j = 0; j < ACH; ++j) {
        const int ci = tid + j * 256, row = ci >> 2, c8 = (ci & 3) * 8;
        ra[j] = *(const v8h*)&Az[(size_t)(m0 + row) * lda + kn + c8];
      }
#pragma unroll
      for (int j = 0; j < BCH; ++j) {
        const int ci = tid + j * 256, row = ci >> 2, c8 = (ci & 3) * 8;
        rb[j] = *(const v8h*)&Bz[(size_t)(n0 + row) * ldb + kn + c8];
      }
    }

    // ---- fragments per ISA VGPR layouts ----
    // A 16x32 f16: group0 holds K0-7 then K16-23; group1 K8-15 then K24-31.
    // B 32x16 f16: group0 holds K0-7 then K16-23; group1 K8-15 then K24-31 (col = lr).
    union Frag { v16h v; v8h h[2]; };
    Frag a[TM], b[TN];
#pragma unroll
    for (int i = 0; i < TM; ++i) {
      const _Float16* p = &sA[cur][(size_t)(wm * TM * 16 + i * 16 + lr) * SLD + lg * 8];
      a[i].h[0] = *(const v8h*)p;
      a[i].h[1] = *(const v8h*)(p + 16);
    }
#pragma unroll
    for (int j = 0; j < TN; ++j) {
      const _Float16* p = &sB[cur][(size_t)(wn * TN * 16 + j * 16 + lr) * SLD + lg * 8];
      b[j].h[0] = *(const v8h*)p;
      b[j].h[1] = *(const v8h*)(p + 16);
    }
#pragma unroll
    for (int i = 0; i < TM; ++i)
#pragma unroll
      for (int j = 0; j < TN; ++j)
        acc[i][j] = __builtin_amdgcn_wmma_f32_16x16x32_f16(
            false, a[i].v, false, b[j].v, (short)0, acc[i][j], false, false);

    // ---- store prefetched tile into the other buffer ----
    if (hasNext) {
      const int nxt = cur ^ 1;
#pragma unroll
      for (int j = 0; j < ACH; ++j) {
        const int ci = tid + j * 256, row = ci >> 2, c8 = (ci & 3) * 8;
        *(v8h*)&sA[nxt][row * SLD + c8] = ra[j];
      }
#pragma unroll
      for (int j = 0; j < BCH; ++j) {
        const int ci = tid + j * 256, row = ci >> 2, c8 = (ci & 3) * 8;
        *(v8h*)&sB[nxt][row * SLD + c8] = rb[j];
      }
    }
    cur ^= 1;
  }

  // ---- epilogue: scale + bias + relu, generic scatter ----
  // C/D f32 layout: VGPR i -> lanes 0-15: (M=i, N=lane); lanes 16-31: (M=8+i)
  const size_t zoff = (size_t)(z / zsplit) * pz0 + (size_t)(z % zsplit) * pz1;
#pragma unroll
  for (int mi = 0; mi < TM; ++mi) {
#pragma unroll
    for (int ni = 0; ni < TN; ++ni) {
      const int n = n0 + wn * TN * 16 + ni * 16 + lr;
      const size_t noff = (size_t)(n / nsplit) * pn0 + (size_t)(n % nsplit) * pn1;
      const float bvl = bias ? bias[n] : 0.0f;
#pragma unroll
      for (int i = 0; i < 8; ++i) {
        const int m = m0 + wm * TM * 16 + mi * 16 + lg * 8 + i;
        float val = acc[mi][ni][i] * scale + bvl;
        if (relu) val = fmaxf(val, 0.0f);
        C[zoff + (size_t)(m / msplit) * pm0 + (size_t)(m % msplit) * pm1 + noff]
            = (OutT)val;
      }
    }
  }
}

// --------------------- row softmax, in-place on f16 -------------------------
__global__ __launch_bounds__(256)
void softmax_inplace_f16_kernel(_Float16* __restrict__ data, int ncols) {
  __shared__ float red[256];
  _Float16* p = data + (size_t)blockIdx.x * ncols;
  const int t = threadIdx.x;

  float mx = -3.0e30f;
  for (int c = t; c < ncols; c += 256) mx = fmaxf(mx, (float)p[c]);
  red[t] = mx; __syncthreads();
  for (int s = 128; s > 0; s >>= 1) {
    if (t < s) red[t] = fmaxf(red[t], red[t + s]);
    __syncthreads();
  }
  const float rowmax = red[0];
  __syncthreads();

  float sum = 0.0f;
  for (int c = t; c < ncols; c += 256) {
    float e = __expf((float)p[c] - rowmax);
    p[c] = (_Float16)e;
    sum += e;
  }
  red[t] = sum; __syncthreads();
  for (int s = 128; s > 0; s >>= 1) {
    if (t < s) red[t] += red[t + s];
    __syncthreads();
  }
  const float inv = 1.0f / red[0];
  for (int c = t; c < ncols; c += 256)
    p[c] = (_Float16)((float)p[c] * inv);
}

// ------------- residual add + layernorm (f32 out, optional f16 out) --------
__global__ __launch_bounds__(256)
void add_ln_kernel(const float* __restrict__ a, const float* __restrict__ b,
                   const float* __restrict__ g, const float* __restrict__ be,
                   float* __restrict__ out32, _Float16* __restrict__ out16,
                   int ncols) {
  __shared__ float redS[256];
  __shared__ float redQ[256];
  const size_t row = blockIdx.x;
  const float* pa = a + row * ncols;
  const float* pb = b + row * ncols;
  const int t = threadIdx.x;

  float s = 0.0f, q = 0.0f;
  for (int c = t; c < ncols; c += 256) {
    float v = pa[c] + pb[c];
    s += v; q += v * v;
  }
  redS[t] = s; redQ[t] = q; __syncthreads();
  for (int st = 128; st > 0; st >>= 1) {
    if (t < st) { redS[t] += redS[t + st]; redQ[t] += redQ[t + st]; }
    __syncthreads();
  }
  const float mu   = redS[0] / (float)ncols;
  const float var  = redQ[0] / (float)ncols - mu * mu;
  const float rstd = rsqrtf(var + 1e-5f);

  for (int c = t; c < ncols; c += 256) {
    float v = pa[c] + pb[c];
    float y = (v - mu) * rstd * g[c] + be[c];
    out32[row * (size_t)ncols + c] = y;
    if (out16) out16[row * (size_t)ncols + c] = (_Float16)y;
  }
}

// ---------------------------------------------------------------------------
extern "C" void kernel_launch(void* const* d_in, const int* in_sizes, int n_in,
                              void* d_out, int out_size, void* d_ws, size_t ws_size,
                              hipStream_t stream) {
  (void)in_sizes; (void)n_in; (void)out_size; (void)ws_size;

  const float* x   = (const float*)d_in[0];
  const float* Wq  = (const float*)d_in[1];
  const float* bq  = (const float*)d_in[2];
  const float* Wk  = (const float*)d_in[3];
  const float* bk  = (const float*)d_in[4];
  const float* Wv  = (const float*)d_in[5];
  const float* bv  = (const float*)d_in[6];
  const float* Wo  = (const float*)d_in[7];
  const float* bo  = (const float*)d_in[8];
  const float* W1  = (const float*)d_in[9];
  const float* b1  = (const float*)d_in[10];
  const float* W2  = (const float*)d_in[11];
  const float* b2  = (const float*)d_in[12];
  const float* g1  = (const float*)d_in[13];
  const float* be1 = (const float*)d_in[14];
  const float* g2  = (const float*)d_in[15];
  const float* be2 = (const float*)d_in[16];
  float* out = (float*)d_out;

  // ----- workspace carve (256B aligned) -----
  char* ws = (char*)d_ws;
  size_t off = 0;
  auto take = [&](size_t bytes) -> void* {
    void* p = ws + off;
    off = (off + bytes + 255) & ~(size_t)255;
    return p;
  };
  _Float16* xh   = (_Float16*)take((size_t)MROWS * DM * 2);
  _Float16* wqt  = (_Float16*)take((size_t)DM * DM * 2);
  _Float16* wkt  = (_Float16*)take((size_t)DM * DM * 2);
  _Float16* wvt  = (_Float16*)take((size_t)DM * DM * 2);
  _Float16* wot  = (_Float16*)take((size_t)DM * DM * 2);
  _Float16* w1t  = (_Float16*)take((size_t)DM * DFF * 2);   // [DFF][DM]
  _Float16* w2t  = (_Float16*)take((size_t)DFF * DM * 2);   // [DM][DFF]
  _Float16* qh   = (_Float16*)take((size_t)MROWS * DM * 2); // [B,H,S,dk]
  _Float16* kh   = (_Float16*)take((size_t)MROWS * DM * 2); // [B,H,S,dk]
  _Float16* vth  = (_Float16*)take((size_t)MROWS * DM * 2); // [B,H,dk,S]
  _Float16* sc   = (_Float16*)take((size_t)NBZ * SEQ * SEQ * 2); // scores/attn
  _Float16* ctxh = (_Float16*)take((size_t)MROWS * DM * 2); // [B,S,H*dk]
  float*    aout = (float*)take((size_t)MROWS * DM * 4);
  float*    h32  = (float*)take((size_t)MROWS * DM * 4);
  _Float16* h16  = (_Float16*)take((size_t)MROWS * DM * 2);
  _Float16* relh = (_Float16*)take((size_t)MROWS * DFF * 2);
  float*    ff2o = (float*)take((size_t)MROWS * DM * 4);

  const int BIG = 1 << 30;

  // Tile configs: big = 128x128 block (waves 2x4, each 4x2 WMMA tiles);
  //               nar = 128x64  block (waves 4x2, each 2x2 WMMA tiles).
  #define GEMM_BIG(T) gemm_bt_kernel<T, 4, 2, 2, 4>
  #define GEMM_NAR(T) gemm_bt_kernel<T, 2, 2, 4, 2>

  // ----- precision conversion / weight transposes -----
  cvt_f32_f16_kernel<<<(MROWS * DM) / 256, 256, 0, stream>>>(x, xh, MROWS * DM);
  transpose_f32_f16_kernel<<<(DM * DM) / 256, 256, 0, stream>>>(Wq, wqt, DM, DM, DM * DM);
  transpose_f32_f16_kernel<<<(DM * DM) / 256, 256, 0, stream>>>(Wk, wkt, DM, DM, DM * DM);
  transpose_f32_f16_kernel<<<(DM * DM) / 256, 256, 0, stream>>>(Wv, wvt, DM, DM, DM * DM);
  transpose_f32_f16_kernel<<<(DM * DM) / 256, 256, 0, stream>>>(Wo, wot, DM, DM, DM * DM);
  transpose_f32_f16_kernel<<<(DM * DFF) / 256, 256, 0, stream>>>(W1, w1t, DM, DFF, DM * DFF);
  transpose_f32_f16_kernel<<<(DFF * DM) / 256, 256, 0, stream>>>(W2, w2t, DFF, DM, DFF * DM);

  // ----- Q projection: out q[b][h][s][d] -----
  GEMM_BIG(_Float16)<<<dim3(DM / 128, MROWS / 128, 1), 256, 0, stream>>>(
      xh, 0, DM, wqt, 0, DM, qh, BIG, 0, 0,
      DM, bq, 1.0f, 0,
      SEQ, (size_t)NH * SEQ * DKH, (size_t)DKH,      // m=(b,s): b,s strides
      DKH, (size_t)SEQ * DKH, (size_t)1);            // n=(h,d): h,d strides

  // ----- K projection: out k[b][h][s][d] -----
  GEMM_BIG(_Float16)<<<dim3(DM / 128, MROWS / 128, 1), 256, 0, stream>>>(
      xh, 0, DM, wkt, 0, DM, kh, BIG, 0, 0,
      DM, bk, 1.0f, 0,
      SEQ, (size_t)NH * SEQ * DKH, (size_t)DKH,
      DKH, (size_t)SEQ * DKH, (size_t)1);

  // ----- V projection, transposed per head: out vt[b][h][d][s] -----
  GEMM_BIG(_Float16)<<<dim3(DM / 128, MROWS / 128, 1), 256, 0, stream>>>(
      xh, 0, DM, wvt, 0, DM, vth, BIG, 0, 0,
      DM, bv, 1.0f, 0,
      SEQ, (size_t)NH * DKH * SEQ, (size_t)1,        // m=(b,s): s innermost
      DKH, (size_t)DKH * SEQ, (size_t)SEQ);          // n=(h,d): d stride = SEQ

  // ----- scores = (Q @ K^T) / sqrt(dk), batched over z = b*H+h -----
  GEMM_BIG(_Float16)<<<dim3(SEQ / 128, SEQ / 128, NBZ), 256, 0, stream>>>(
      qh, (size_t)SEQ * DKH, DKH, kh, (size_t)SEQ * DKH, DKH,
      sc, BIG, 0, (size_t)SEQ * SEQ,
      DKH, nullptr, 0.125f, 0,
      SEQ, 0, (size_t)SEQ,
      128, (size_t)128, (size_t)1);

  // ----- softmax rows, in place -----
  softmax_inplace_f16_kernel<<<NBZ * SEQ, 256, 0, stream>>>(sc, SEQ);

  // ----- ctx = attn @ V -> ctx[b][s][h*64+d], batched z=(b,h) -----
  GEMM_NAR(_Float16)<<<dim3(DKH / 64, SEQ / 128, NBZ), 256, 0, stream>>>(
      sc, (size_t)SEQ * SEQ, SEQ, vth, (size_t)DKH * SEQ, SEQ,
      ctxh, NH, (size_t)SEQ * DM, (size_t)DKH,       // zoff = b*S*DM + h*64
      SEQ, nullptr, 1.0f, 0,
      SEQ, 0, (size_t)DM,                            // m=s: stride DM
      DKH, 0, (size_t)1);                            // n=d: stride 1

  // ----- attn_out = ctx @ Wo + bo (f32, plain [MROWS][DM]) -----
  GEMM_BIG(float)<<<dim3(DM / 128, MROWS / 128, 1), 256, 0, stream>>>(
      ctxh, 0, DM, wot, 0, DM, aout, BIG, 0, 0,
      DM, bo, 1.0f, 0,
      MROWS, 0, (size_t)DM,
      128, (size_t)128, (size_t)1);

  // ----- h = LN1(x + attn_out), f32 + f16 copies -----
  add_ln_kernel<<<MROWS, 256, 0, stream>>>(x, aout, g1, be1, h32, h16, DM);

  // ----- ff hidden = relu(h @ W1 + b1), f16 [MROWS][DFF] -----
  GEMM_BIG(_Float16)<<<dim3(DFF / 128, MROWS / 128, 1), 256, 0, stream>>>(
      h16, 0, DM, w1t, 0, DM, relh, BIG, 0, 0,
      DM, b1, 1.0f, 1,
      MROWS, 0, (size_t)DFF,
      128, (size_t)128, (size_t)1);

  // ----- ff out = hidden @ W2 + b2, f32 [MROWS][DM] -----
  GEMM_BIG(float)<<<dim3(DM / 128, MROWS / 128, 1), 256, 0, stream>>>(
      relh, 0, DFF, w2t, 0, DFF, ff2o, BIG, 0, 0,
      DFF, b2, 1.0f, 0,
      MROWS, 0, (size_t)DM,
      128, (size_t)128, (size_t)1);

  // ----- final output = LN2(h + ff) -----
  add_ln_kernel<<<MROWS, 256, 0, stream>>>(h32, ff2o, g2, be2, out, nullptr, DM);
}
